// SSM_4243427689264
// MI455X (gfx1250) — matrix operand order repaired and checked by reading, not compiled, not run on hardware
//
#include <hip/hip_runtime.h>
#include <hip/hip_bf16.h>
#include <math.h>

// Problem dims (fixed by the reference)
#define BSZ   8
#define LSEQ  2048
#define DMM   1024
#define DHH   1024
#define MROWS (BSZ * LSEQ)          // 16384
#define NCHUNK 16
#define CHUNK  (LSEQ / NCHUNK)      // 128

typedef __bf16 bf16_t;
typedef bf16_t v16bf __attribute__((ext_vector_type(16)));
typedef bf16_t v8bf  __attribute__((ext_vector_type(8)));
typedef bf16_t v4bf  __attribute__((ext_vector_type(4)));
typedef float  v8f   __attribute__((ext_vector_type(8)));

// ---------- fp32 -> bf16 (round-to-nearest-even); only used in mem-bound passes ----------
__device__ __forceinline__ bf16_t f2bf(float f) {
    unsigned u = __builtin_bit_cast(unsigned, f);
    u += 0x7FFFu + ((u >> 16) & 1u);
    return __builtin_bit_cast(bf16_t, (unsigned short)(u >> 16));
}

__device__ __forceinline__ v8f wmma_bf16(v16bf a, v16bf b, v8f c) {
    return __builtin_amdgcn_wmma_f32_16x16x32_bf16(
        false, a, false, b, (short)0, c, false, false);
}

// ---------- CDNA5 async global->LDS (ASYNCcnt-tracked, bypasses VGPRs) ----------
// vdst VGPR = wave-relative LDS byte address.
__device__ __forceinline__ void async_b128(unsigned ldsOff, const void* g) {
    asm volatile("global_load_async_to_lds_b128 %0, %1, off"
                 :: "v"(ldsOff), "v"(g) : "memory");
}
__device__ __forceinline__ void wait_async0() {
    asm volatile("s_wait_asynccnt 0x0" ::: "memory");
}
__device__ __forceinline__ unsigned lds_off(const void* p) {
    return (unsigned)(size_t)p;      // low 32 bits of generic LDS addr = LDS byte address
}

// ---------- WMMA operand fetch from an LDS tile (row-major [rows][32] bf16) ----------
// A 16x32: lanes 0-15 K={0..7,16..23}, lanes 16-31 K={8..15,24..31}
__device__ __forceinline__ v16bf lds_a(const bf16_t* S, int rowOff, int lane) {
    const bf16_t* p = S + (unsigned)(rowOff + (lane & 15)) * 32u + 8u * (lane >> 4);
    v8bf c0 = *(const v8bf*)p;
    v8bf c1 = *(const v8bf*)(p + 16);
    return __builtin_shufflevector(c0, c1, 0,1,2,3,4,5,6,7,8,9,10,11,12,13,14,15);
}
// B 32x16: lanes 0-15 K={0..7,16..23}, lanes 16-31 K={8..15,24..31}
__device__ __forceinline__ v16bf lds_b(const bf16_t* S, int nOff, int lane) {
    const bf16_t* p = S + (unsigned)(nOff + (lane & 15)) * 32u + 8u * (lane >> 4);
    v8bf c0 = *(const v8bf*)p;
    v8bf c1 = *(const v8bf*)(p + 16);
    return __builtin_shufflevector(c0, c1, 0,1,2,3,4,5,6,7,8,9,10,11,12,13,14,15);
}

// Stage one [rows x 32] bf16 panel (rows*64 bytes) with `rows/64` chunks per thread.
// Each thread covers 16B: row = t>>2 (+64*q), k-chunk = 8*(t&3).
__device__ __forceinline__ void stage_panel(const bf16_t* __restrict__ G, int row0,
                                            int k0, bf16_t* sdst, unsigned t, int qcnt) {
    const unsigned sbase = lds_off(sdst);
#pragma unroll
    for (int q = 0; q < 4; ++q) {
        if (q >= qcnt) break;
        const unsigned r = (t >> 2) + 64u * q;
        const bf16_t* g = G + (size_t)(row0 + r) * 1024u + (unsigned)k0 + 8u * (t & 3);
        async_b128(sbase + r * 64u + 16u * (t & 3), g);
    }
}

// =====================================================================
// fp32 -> bf16 bulk conversion (memory bound; n multiple of 4)
// =====================================================================
__global__ __launch_bounds__(256)
void cvt_f32_to_bf16(const float* __restrict__ src, bf16_t* __restrict__ dst, int n4) {
    const int i = blockIdx.x * blockDim.x + threadIdx.x;
    if (i >= n4) return;
    const float4 v = ((const float4*)src)[i];
    v4bf o;
    o[0] = f2bf(v.x); o[1] = f2bf(v.y); o[2] = f2bf(v.z); o[3] = f2bf(v.w);
    ((v4bf*)dst)[i] = o;
}

// =====================================================================
// Phase 1: dual GEMM  dRaw = x@dw^T, Bx = x@Bw^T  (+ fused SSM elementwise)
// Block tile 64(M) x 256(N), K-step 32, double-buffered async LDS staging.
// 8 waves = 2(M) x 4(N); wave tile 32 x 64.
// LDS buffer: [A 64x32 | dw 256x32 | Bw 256x32] = 18432 bf16 = 36 KB, x2 buffers.
// =====================================================================
#define P1_A   0
#define P1_W0  2048
#define P1_W1  (2048 + 8192)
#define P1_BUF 18432

__global__ __launch_bounds__(256)
void ssm_phase1(const bf16_t* __restrict__ xbf,
                const bf16_t* __restrict__ dwbf, const float* __restrict__ db,
                const bf16_t* __restrict__ Bwbf, const float* __restrict__ Bb,
                const float* __restrict__ Avec,
                float* __restrict__ Abar, float* __restrict__ Bbar) {
    __shared__ __align__(32) bf16_t smem[2 * P1_BUF];

    const unsigned t = threadIdx.x;
    const int lane = t & 31;
    const int wave = t >> 5;
    const int mb = blockIdx.y * 64;               // block M origin
    const int nb = blockIdx.x * 256;              // block N origin
    const int wm = (wave >> 2) * 32;              // wave M offset in block
    const int wn = (wave & 3) * 64;               // wave N offset in block

    v8f accD[2][4] = {}, accB[2][4] = {};

    // Prologue: stage k0 = 0 into buffer 0
    stage_panel(xbf,  mb, 0, smem + P1_A,  t, 1);
    stage_panel(dwbf, nb, 0, smem + P1_W0, t, 4);
    stage_panel(Bwbf, nb, 0, smem + P1_W1, t, 4);
    wait_async0();
    __syncthreads();

    int buf = 0;
    for (int k0 = 0; k0 < DMM; k0 += 32) {
        bf16_t* nxt = smem + (buf ^ 1) * P1_BUF;
        if (k0 + 32 < DMM) {                      // prefetch next K-step
            stage_panel(xbf,  mb, k0 + 32, nxt + P1_A,  t, 1);
            stage_panel(dwbf, nb, k0 + 32, nxt + P1_W0, t, 4);
            stage_panel(Bwbf, nb, k0 + 32, nxt + P1_W1, t, 4);
        }
        const bf16_t* sb = smem + buf * P1_BUF;
        const v16bf a0 = lds_a(sb + P1_A, wm,      lane);
        const v16bf a1 = lds_a(sb + P1_A, wm + 16, lane);
#pragma unroll
        for (int j = 0; j < 4; ++j) {
            const v16bf bd = lds_b(sb + P1_W0, wn + 16 * j, lane);
            accD[0][j] = wmma_bf16(a0, bd, accD[0][j]);
            accD[1][j] = wmma_bf16(a1, bd, accD[1][j]);
            const v16bf bb = lds_b(sb + P1_W1, wn + 16 * j, lane);
            accB[0][j] = wmma_bf16(a0, bb, accB[0][j]);
            accB[1][j] = wmma_bf16(a1, bb, accB[1][j]);
        }
        wait_async0();
        __syncthreads();
        buf ^= 1;
    }

    // C/D layout: VGPR r, lanes 0-15 -> (M=r, N=lane); lanes 16-31 -> (M=r+8, N=lane-16)
    const int half = lane >> 4, nl = lane & 15;
#pragma unroll
    for (int j = 0; j < 4; ++j) {
        const int n = nb + wn + 16 * j + nl;
        const float dbv = db[n], bbv = Bb[n], av = Avec[n];
#pragma unroll
        for (int i = 0; i < 2; ++i) {
#pragma unroll
            for (int r = 0; r < 8; ++r) {
                const int m = mb + wm + 16 * i + r + 8 * half;
                const float draw  = accD[i][j][r] + dbv;
                const float delta = (draw > 20.0f) ? draw : log1pf(__expf(draw));
                const float abar  = __expf(delta * av);
                const float bbar  = delta * (accB[i][j][r] + bbv);
                const size_t idx = (size_t)m * DHH + n;
                Abar[idx] = abar;
                Bbar[idx] = bbar;
            }
        }
    }
}

// =====================================================================
// Chunked parallel scan of  h_t = A_bar_t * h_{t-1} + B_bar_t  over L.
// =====================================================================
__global__ __launch_bounds__(256)
void scan_pass1(const float* __restrict__ Abar, const float* __restrict__ Bbar,
                float* __restrict__ P, float* __restrict__ F) {
    const size_t tid = (size_t)blockIdx.x * blockDim.x + threadIdx.x;
    const int hh = (int)(tid & (DHH - 1));
    const int c  = (int)((tid >> 10) & (NCHUNK - 1));
    const int b  = (int)(tid >> 14);
    size_t base = ((size_t)b * LSEQ + (size_t)c * CHUNK) * DHH + hh;
    float p = 1.0f, f = 0.0f;
#pragma unroll 4
    for (int tt = 0; tt < CHUNK; ++tt) {
        const float a  = Abar[base + (size_t)tt * DHH];
        const float bb = Bbar[base + (size_t)tt * DHH];
        p *= a;
        f  = f * a + bb;
    }
    P[tid] = p;
    F[tid] = f;
}

__global__ __launch_bounds__(256)
void scan_pass2(const float* __restrict__ P, float* __restrict__ F) {
    const int tid = blockIdx.x * blockDim.x + threadIdx.x;   // < BSZ*DHH
    const int hh = tid & (DHH - 1);
    const int b  = tid >> 10;
    float state = 0.0f;
    for (int c = 0; c < NCHUNK; ++c) {
        const size_t i = ((size_t)b * NCHUNK + c) * DHH + hh;
        const float p = P[i], f = F[i];
        F[i] = state;               // carry-in for chunk c
        state = f + p * state;
    }
}

__global__ __launch_bounds__(256)
void scan_pass3(const float* __restrict__ Abar, const float* __restrict__ Bbar,
                const float* __restrict__ F, bf16_t* __restrict__ hbf) {
    const size_t tid = (size_t)blockIdx.x * blockDim.x + threadIdx.x;
    const int hh = (int)(tid & (DHH - 1));
    const int c  = (int)((tid >> 10) & (NCHUNK - 1));
    const int b  = (int)(tid >> 14);
    size_t base = ((size_t)b * LSEQ + (size_t)c * CHUNK) * DHH + hh;
    float state = F[tid];
#pragma unroll 4
    for (int tt = 0; tt < CHUNK; ++tt) {
        const size_t idx = base + (size_t)tt * DHH;
        state = state * Abar[idx] + Bbar[idx];
        hbf[idx] = f2bf(state);
    }
}

// =====================================================================
// Phase 3: out = h@Cw^T + x@Dw^T + Cb + Db  (two GEMMs into one accumulator)
// LDS buffer: [Ah 64x32 | Ax 64x32 | Cw 256x32 | Dw 256x32] = 20480 bf16 = 40 KB, x2.
// =====================================================================
#define P3_AH  0
#define P3_AX  2048
#define P3_W0  4096
#define P3_W1  (4096 + 8192)
#define P3_BUF 20480

__global__ __launch_bounds__(256)
void ssm_phase3(const bf16_t* __restrict__ hbf,
                const bf16_t* __restrict__ Cwbf, const float* __restrict__ Cb,
                const bf16_t* __restrict__ xbf,
                const bf16_t* __restrict__ Dwbf, const float* __restrict__ Db,
                float* __restrict__ out) {
    __shared__ __align__(32) bf16_t smem[2 * P3_BUF];

    const unsigned t = threadIdx.x;
    const int lane = t & 31;
    const int wave = t >> 5;
    const int mb = blockIdx.y * 64;
    const int nb = blockIdx.x * 256;
    const int wm = (wave >> 2) * 32;
    const int wn = (wave & 3) * 64;

    v8f acc[2][4] = {};

    stage_panel(hbf,  mb, 0, smem + P3_AH, t, 1);
    stage_panel(xbf,  mb, 0, smem + P3_AX, t, 1);
    stage_panel(Cwbf, nb, 0, smem + P3_W0, t, 4);
    stage_panel(Dwbf, nb, 0, smem + P3_W1, t, 4);
    wait_async0();
    __syncthreads();

    int buf = 0;
    for (int k0 = 0; k0 < DHH; k0 += 32) {
        bf16_t* nxt = smem + (buf ^ 1) * P3_BUF;
        if (k0 + 32 < DHH) {
            stage_panel(hbf,  mb, k0 + 32, nxt + P3_AH, t, 1);
            stage_panel(xbf,  mb, k0 + 32, nxt + P3_AX, t, 1);
            stage_panel(Cwbf, nb, k0 + 32, nxt + P3_W0, t, 4);
            stage_panel(Dwbf, nb, k0 + 32, nxt + P3_W1, t, 4);
        }
        const bf16_t* sb = smem + buf * P3_BUF;
        const v16bf ah0 = lds_a(sb + P3_AH, wm,      lane);
        const v16bf ah1 = lds_a(sb + P3_AH, wm + 16, lane);
        const v16bf ax0 = lds_a(sb + P3_AX, wm,      lane);
        const v16bf ax1 = lds_a(sb + P3_AX, wm + 16, lane);
#pragma unroll
        for (int j = 0; j < 4; ++j) {
            const v16bf bc = lds_b(sb + P3_W0, wn + 16 * j, lane);
            acc[0][j] = wmma_bf16(ah0, bc, acc[0][j]);
            acc[1][j] = wmma_bf16(ah1, bc, acc[1][j]);
            const v16bf bd = lds_b(sb + P3_W1, wn + 16 * j, lane);
            acc[0][j] = wmma_bf16(ax0, bd, acc[0][j]);
            acc[1][j] = wmma_bf16(ax1, bd, acc[1][j]);
        }
        wait_async0();
        __syncthreads();
        buf ^= 1;
    }

    const int half = lane >> 4, nl = lane & 15;
#pragma unroll
    for (int j = 0; j < 4; ++j) {
        const int n = nb + wn + 16 * j + nl;
        const float bias = Cb[n] + Db[n];
#pragma unroll
        for (int i = 0; i < 2; ++i) {
#pragma unroll
            for (int r = 0; r < 8; ++r) {
                const int m = mb + wm + 16 * i + r + 8 * half;
                out[(size_t)m * DMM + n] = acc[i][j][r] + bias;
            }
        }
    }
}

extern "C" void kernel_launch(void* const* d_in, const int* in_sizes, int n_in,
                              void* d_out, int out_size, void* d_ws, size_t ws_size,
                              hipStream_t stream) {
    // setup_inputs order: x, A, Bw, Bb, Cw, Cb, Dw, Db, dw, db  (all fp32)
    const float* x    = (const float*)d_in[0];
    const float* Avec = (const float*)d_in[1];
    const float* Bw   = (const float*)d_in[2];
    const float* Bb   = (const float*)d_in[3];
    const float* Cw   = (const float*)d_in[4];
    const float* Cb   = (const float*)d_in[5];
    const float* Dw   = (const float*)d_in[6];
    const float* Db   = (const float*)d_in[7];
    const float* dw   = (const float*)d_in[8];
    const float* db   = (const float*)d_in[9];
    float* out = (float*)d_out;

    const size_t NX = (size_t)MROWS * DHH;   // 16M elements
    const size_t NW = (size_t)DHH * DMM;     // 1M elements per weight matrix
    const size_t NC = (size_t)BSZ * NCHUNK * DHH;

    // Workspace carving
    float*  Abar   = (float*)d_ws;                 // 64 MB
    float*  Bbar   = Abar + NX;                    // 64 MB
    float*  carryP = Bbar + NX;                    // 512 KB
    float*  carryF = carryP + NC;                  // 512 KB
    bf16_t* xbf    = (bf16_t*)(carryF + NC);       // 32 MB
    bf16_t* hbf    = xbf + NX;                     // 32 MB
    bf16_t* dwbf   = hbf + NX;                     // 2 MB each
    bf16_t* Bwbf   = dwbf + NW;
    bf16_t* Cwbf   = Bwbf + NW;
    bf16_t* Dwbf   = Cwbf + NW;

    const dim3 blk(256);

    // Bulk fp32->bf16 conversions (memory bound)
    cvt_f32_to_bf16<<<(int)(NX / 4 / 256), blk, 0, stream>>>(x,  xbf,  (int)(NX / 4));
    cvt_f32_to_bf16<<<(int)(NW / 4 / 256), blk, 0, stream>>>(dw, dwbf, (int)(NW / 4));
    cvt_f32_to_bf16<<<(int)(NW / 4 / 256), blk, 0, stream>>>(Bw, Bwbf, (int)(NW / 4));
    cvt_f32_to_bf16<<<(int)(NW / 4 / 256), blk, 0, stream>>>(Cw, Cwbf, (int)(NW / 4));
    cvt_f32_to_bf16<<<(int)(NW / 4 / 256), blk, 0, stream>>>(Dw, Dwbf, (int)(NW / 4));

    const dim3 gemm_grid(DHH / 256, MROWS / 64);   // (4, 256)

    ssm_phase1<<<gemm_grid, blk, 0, stream>>>(xbf, dwbf, db, Bwbf, Bb, Avec, Abar, Bbar);

    const int nscan = (int)(BSZ * NCHUNK * DHH);   // 131072
    scan_pass1<<<nscan / 256, blk, 0, stream>>>(Abar, Bbar, carryP, carryF);
    scan_pass2<<<(BSZ * DHH) / 256, blk, 0, stream>>>(carryP, carryF);
    scan_pass3<<<nscan / 256, blk, 0, stream>>>(Abar, Bbar, carryF, hbf);

    ssm_phase3<<<gemm_grid, blk, 0, stream>>>(hbf, Cwbf, Cb, xbf, Dwbf, Db, out);
}
